// GATS_42296837931448
// MI455X (gfx1250) — matrix operand order repaired and checked by reading, compiled, not run on hardware
//
#include <hip/hip_runtime.h>

#define HEADS 8
#define CH 64
#define NEG_SLOPE 0.2f

typedef __attribute__((ext_vector_type(16))) _Float16 v16h;
typedef __attribute__((ext_vector_type(8)))  float    v8f;

// Monotone mapping float -> uint so float max == uint max (handles negatives).
__device__ __forceinline__ unsigned fmax_key(float f) {
    unsigned u = __float_as_uint(f);
    return (u & 0x80000000u) ? ~u : (u | 0x80000000u);
}
__device__ __forceinline__ float fmax_unkey(unsigned k) {
    return (k & 0x80000000u) ? __uint_as_float(k ^ 0x80000000u)
                             : __uint_as_float(~k);
}

// ---------------------------------------------------------------------------
// Kernel 1: h = logits @ gat_w via v_wmma_f32_16x16x32_f16 (K=64 -> 2 steps).
// Fused: a_src = h*att_src, a_dst = h*att_dst.
// One wave per 16-row tile. gat_w [64x8] staged zero-padded in LDS once per
// block; D-matrix is transposed through a per-wave LDS stage so the global
// writeback is 3 coalesced b128 stores per lane (no per-row exec branches).
// ---------------------------------------------------------------------------
__global__ void gat_project_wmma(const float* __restrict__ logits,
                                 const float* __restrict__ gat_w,
                                 const float* __restrict__ att_src,
                                 const float* __restrict__ att_dst,
                                 float* __restrict__ hbuf,
                                 float* __restrict__ asrc,
                                 float* __restrict__ adst,
                                 int n_nodes, int n_tiles) {
    // Padded B: padB[col*64 + k] = (col < 8) ? gat_w[k*8 + col] : 0
    __shared__ __align__(32) _Float16 padB[16 * CH];     // 2 KB
    __shared__ __align__(16) float    stage[8][16 * HEADS]; // 4 KB, per-wave slots

    for (int i = threadIdx.x * 4; i < 16 * CH; i += blockDim.x * 4) {
        const int c = i >> 6;          // column 0..15 (uniform across the 4)
        const int k = i & 63;
        const bool valid = (c < HEADS);
        #pragma unroll
        for (int j = 0; j < 4; ++j) {
            float w = gat_w[(k + j) * HEADS + (c & (HEADS - 1))]; // always in-bounds
            padB[i + j] = valid ? (_Float16)w : (_Float16)0.0f;   // v_cndmask, no branch
        }
    }
    __syncthreads();

    const int wave = (int)((blockIdx.x * blockDim.x + threadIdx.x) >> 5);
    if (wave >= n_tiles) return;                 // uniform per wave: EXEC stays full
    const int lane      = threadIdx.x & 31;
    const int wslot     = (threadIdx.x >> 5) & 7;
    const int tile_base = wave * 16;

    const int col      = lane & 15;              // A row (M) / B,C,D column (N)
    const int half_sel = lane >> 4;              // which K-half group this lane owns
    const int koff     = half_sel * 8;           // A: K in {koff..koff+7, 16+koff..}

    int row = tile_base + col;
    if (row >= n_nodes) row = n_nodes - 1;       // dup-read tail rows; stores guarded
    const float* rp = logits + (size_t)row * CH;

    v8f c = {};
    #pragma unroll
    for (int k0 = 0; k0 < CH; k0 += 32) {
        // ---- A: 16x32 f16, ISA A-layout: lane<16 -> K {0..7,16..23}; lane>=16 -> {8..15,24..31}
        const float4* pa0 = (const float4*)(rp + k0 + koff);
        const float4* pa1 = (const float4*)(rp + k0 + 16 + koff);
        float4 f0 = pa0[0], f1 = pa0[1];
        float4 f2 = pa1[0], f3 = pa1[1];
        v16h a;
        a[0]  = (_Float16)f0.x; a[1]  = (_Float16)f0.y; a[2]  = (_Float16)f0.z; a[3]  = (_Float16)f0.w;
        a[4]  = (_Float16)f1.x; a[5]  = (_Float16)f1.y; a[6]  = (_Float16)f1.z; a[7]  = (_Float16)f1.w;
        a[8]  = (_Float16)f2.x; a[9]  = (_Float16)f2.y; a[10] = (_Float16)f2.z; a[11] = (_Float16)f2.w;
        a[12] = (_Float16)f3.x; a[13] = (_Float16)f3.y; a[14] = (_Float16)f3.z; a[15] = (_Float16)f3.w;

        // ---- B: 32x16 f16, same K order as A: lane<16 -> K {0..7,16..23}; lane>=16 -> {8..15,24..31}
        //        of column `col`.
        const int kb = k0 + koff;
        v16h b;
        #pragma unroll
        for (int s = 0; s < 8; ++s) {
            b[s]     = padB[col * CH + kb + s];
            b[s + 8] = padB[col * CH + kb + 16 + s];
        }

        c = __builtin_amdgcn_wmma_f32_16x16x32_f16(false, a, false, b,
                                                   (short)0, c, false, false);
    }

    // ---- D transpose through LDS: C/D layout (VGPR r: lane<16 -> M=r,N=lane;
    //      lane>=16 -> M=8+r,N=lane-16) scattered into [local_node][head].
    if (col < HEADS) {
        #pragma unroll
        for (int r = 0; r < 8; ++r)
            stage[wslot][(half_sel * 8 + r) * HEADS + col] = c[r];
    }
    __builtin_amdgcn_wave_barrier();   // same-wave DS ops are in-order; keep compiler order

    // ---- Coalesced writeback: lane -> (node = tile_base + lane/2, heads 4*(lane&1)..+3)
    const int nloc = lane >> 1;
    const int hgrp = (lane & 1) * 4;
    const int node = tile_base + nloc;
    float4 hv  = *(const float4*)(&stage[wslot][nloc * HEADS + hgrp]);
    float4 ws4 = *(const float4*)(att_src + hgrp);   // unpredicated, always in-bounds
    float4 wd4 = *(const float4*)(att_dst + hgrp);
    if (node < n_nodes) {
        size_t o = (size_t)node * HEADS + hgrp;
        *(float4*)(hbuf + o) = hv;
        *(float4*)(asrc + o) = make_float4(hv.x * ws4.x, hv.y * ws4.y,
                                           hv.z * ws4.z, hv.w * ws4.w);
        *(float4*)(adst + o) = make_float4(hv.x * wd4.x, hv.y * wd4.y,
                                           hv.z * wd4.z, hv.w * wd4.w);
    }
}

// ---------------------------------------------------------------------------
// Kernel 2: vectorized init of reduction buffers (n is a multiple of 4).
// ---------------------------------------------------------------------------
__global__ void gat_init(uint4* __restrict__ mkey4,
                         float4* __restrict__ denom4,
                         float4* __restrict__ accum4, int n4) {
    int i = blockIdx.x * blockDim.x + threadIdx.x;
    if (i >= n4) return;
    mkey4[i]  = make_uint4(0u, 0u, 0u, 0u); // fmax_unkey(0) < any real value
    denom4[i] = make_float4(0.f, 0.f, 0.f, 0.f);
    accum4[i] = make_float4(0.f, 0.f, 0.f, 0.f);
}

// ---------------------------------------------------------------------------
// Kernel 3: edge pass A — per-destination segment max of leaky_relu(a_s+a_d),
// via monotone-uint atomic_max_u32. Items [0,E) are edges, [E,E+N) self-loops.
// ---------------------------------------------------------------------------
__global__ void gat_edge_max(const int* __restrict__ ei,
                             const float* __restrict__ asrc,
                             const float* __restrict__ adst,
                             unsigned* __restrict__ mkey,
                             int n_edges, int n_total) {
    int idx = blockIdx.x * blockDim.x + threadIdx.x;
    if (idx >= n_total) return;
    int s, d;
    if (idx < n_edges) { s = ei[idx]; d = ei[n_edges + idx]; }
    else               { s = d = idx - n_edges; }

    const float4* ps = (const float4*)(asrc + (size_t)s * HEADS);
    const float4* pd = (const float4*)(adst + (size_t)d * HEADS);
    float av[HEADS], dv[HEADS];
    *(float4*)&av[0] = ps[0]; *(float4*)&av[4] = ps[1];
    *(float4*)&dv[0] = pd[0]; *(float4*)&dv[4] = pd[1];

    unsigned* mk = mkey + (size_t)d * HEADS;
    #pragma unroll
    for (int hh = 0; hh < HEADS; ++hh) {
        float e = av[hh] + dv[hh];
        e = (e > 0.0f) ? e : NEG_SLOPE * e;
        atomicMax(&mk[hh], fmax_key(e));
    }
}

// ---------------------------------------------------------------------------
// Kernel 4: edge pass B — ex = exp(e - m[dst]); denom[dst] += ex;
// accum[dst] += ex * h[src].  (alpha normalization deferred: sum(alpha*h) =
// sum(ex*h)/denom, saving a third edge pass.)
// ---------------------------------------------------------------------------
__global__ void gat_edge_sum(const int* __restrict__ ei,
                             const float* __restrict__ asrc,
                             const float* __restrict__ adst,
                             const float* __restrict__ hbuf,
                             const unsigned* __restrict__ mkey,
                             float* __restrict__ denom,
                             float* __restrict__ accum,
                             int n_edges, int n_total) {
    int idx = blockIdx.x * blockDim.x + threadIdx.x;
    if (idx >= n_total) return;
    int s, d;
    if (idx < n_edges) { s = ei[idx]; d = ei[n_edges + idx]; }
    else               { s = d = idx - n_edges; }

    const float4* ps = (const float4*)(asrc + (size_t)s * HEADS);
    const float4* pd = (const float4*)(adst + (size_t)d * HEADS);
    const float4* ph = (const float4*)(hbuf + (size_t)s * HEADS);
    const uint4*  pm = (const uint4*)(mkey + (size_t)d * HEADS);
    float av[HEADS], dv[HEADS], hv[HEADS];
    unsigned mk[HEADS];
    *(float4*)&av[0] = ps[0]; *(float4*)&av[4] = ps[1];
    *(float4*)&dv[0] = pd[0]; *(float4*)&dv[4] = pd[1];
    *(float4*)&hv[0] = ph[0]; *(float4*)&hv[4] = ph[1];
    *(uint4*)&mk[0]  = pm[0]; *(uint4*)&mk[4]  = pm[1];

    float* dn = denom + (size_t)d * HEADS;
    float* ac = accum + (size_t)d * HEADS;
    #pragma unroll
    for (int hh = 0; hh < HEADS; ++hh) {
        float e = av[hh] + dv[hh];
        e = (e > 0.0f) ? e : NEG_SLOPE * e;
        float ex = __expf(e - fmax_unkey(mk[hh]));
        atomicAdd(&dn[hh], ex);
        atomicAdd(&ac[hh], ex * hv[hh]);
    }
}

// ---------------------------------------------------------------------------
// Kernel 5: gat_out = accum/denom + bias; temp = relu(gat_out.lin_w + lin_b)
// + temp_bias; out = logits / temp.
// ---------------------------------------------------------------------------
__global__ void gat_finalize(const float* __restrict__ logits,
                             const float* __restrict__ accum,
                             const float* __restrict__ denom,
                             const float* __restrict__ gat_bias,
                             const float* __restrict__ lin_w,
                             const float* __restrict__ lin_b,
                             const float* __restrict__ temp_bias,
                             float* __restrict__ out, int n_nodes) {
    int i = blockIdx.x * blockDim.x + threadIdx.x;
    if (i >= n_nodes) return;

    const float4* ac4 = (const float4*)(accum + (size_t)i * HEADS);
    const float4* dn4 = (const float4*)(denom + (size_t)i * HEADS);
    float ac[HEADS], dn[HEADS], gb[HEADS], lw[HEADS];
    *(float4*)&ac[0] = ac4[0]; *(float4*)&ac[4] = ac4[1];
    *(float4*)&dn[0] = dn4[0]; *(float4*)&dn[4] = dn4[1];
    *(float4*)&gb[0] = *(const float4*)(gat_bias);
    *(float4*)&gb[4] = *(const float4*)(gat_bias + 4);
    *(float4*)&lw[0] = *(const float4*)(lin_w);
    *(float4*)&lw[4] = *(const float4*)(lin_w + 4);

    float t = 0.0f;
    #pragma unroll
    for (int hh = 0; hh < HEADS; ++hh)
        t += (ac[hh] / dn[hh] + gb[hh]) * lw[hh];
    t += lin_b[0];
    t = (t > 0.0f ? t : 0.0f) + temp_bias[0];
    float inv = 1.0f / t;

    const float4* lp = (const float4*)(logits + (size_t)i * CH);
    float4*       op = (float4*)(out + (size_t)i * CH);
    #pragma unroll
    for (int j = 0; j < CH / 4; ++j) {
        float4 v = lp[j];
        v.x *= inv; v.y *= inv; v.z *= inv; v.w *= inv;
        op[j] = v;
    }
}

extern "C" void kernel_launch(void* const* d_in, const int* in_sizes, int n_in,
                              void* d_out, int out_size, void* d_ws, size_t ws_size,
                              hipStream_t stream) {
    const float* logits    = (const float*)d_in[0];
    const int*   ei        = (const int*)d_in[1];
    const float* gat_w     = (const float*)d_in[2];
    const float* att_src   = (const float*)d_in[3];
    const float* att_dst   = (const float*)d_in[4];
    const float* gat_bias  = (const float*)d_in[5];
    const float* lin_w     = (const float*)d_in[6];
    const float* lin_b     = (const float*)d_in[7];
    const float* temp_bias = (const float*)d_in[8];
    float*       out       = (float*)d_out;

    const int n_nodes = in_sizes[0] / CH;
    const int n_edges = in_sizes[1] / 2;
    const int n_total = n_edges + n_nodes;
    const size_t nh   = (size_t)n_nodes * HEADS;

    // Workspace layout (all 32-bit): h, a_src, a_dst, mkey(u32), denom, accum
    float*    hbuf  = (float*)d_ws;
    float*    asrc  = hbuf + nh;
    float*    adst  = asrc + nh;
    unsigned* mkey  = (unsigned*)(adst + nh);
    float*    denom = (float*)(mkey + nh);
    float*    accum = denom + nh;

    const dim3 blk(256);
    const int n_tiles = (n_nodes + 15) / 16;
    const int nh4 = (int)(nh / 4);

    gat_project_wmma<<<dim3((n_tiles + 7) / 8), blk, 0, stream>>>(
        logits, gat_w, att_src, att_dst, hbuf, asrc, adst, n_nodes, n_tiles);

    gat_init<<<dim3((nh4 + 255) / 256), blk, 0, stream>>>(
        (uint4*)mkey, (float4*)denom, (float4*)accum, nh4);

    gat_edge_max<<<dim3((n_total + 255) / 256), blk, 0, stream>>>(
        ei, asrc, adst, mkey, n_edges, n_total);

    gat_edge_sum<<<dim3((n_total + 255) / 256), blk, 0, stream>>>(
        ei, asrc, adst, hbuf, mkey, denom, accum, n_edges, n_total);

    gat_finalize<<<dim3((n_nodes + 255) / 256), blk, 0, stream>>>(
        logits, accum, denom, gat_bias, lin_w, lin_b, temp_bias, out, n_nodes);
}
